// MultiHeadSelfAttention_78726750536236
// MI455X (gfx1250) — matrix operand order repaired and checked by reading, not compiled, not run on hardware
//
#include <hip/hip_runtime.h>

// ---------------------------------------------------------------------------
// MHA forward (RoPE + causal flash attention) for gfx1250 / MI455X.
// All GEMM flops routed through v_wmma_f32_16x16x32_f16 (f32 accumulate).
// Round 2: fused QKV (12 WMMAs per A-fragment), register double-buffering in
// all k-loops, V/K load hoisting over the softmax VALU block, global prefetch.
// ---------------------------------------------------------------------------

#define DMODEL 768
#define NH     12
#define HD     64
#define SEQL   2048
#define BATCH  4
#define BS     (BATCH * SEQL)

typedef __attribute__((ext_vector_type(8)))  _Float16 v8h;
typedef __attribute__((ext_vector_type(16))) _Float16 v16h;
typedef __attribute__((ext_vector_type(8)))  float    v8f;

// A-fragment (16x32 f16, M x K): this lane holds row (lane&15).
// kbA = (lane>>4)*8 ; element j -> K=2*kbA+j (contiguous, matches ldB).
__device__ __forceinline__ v16h ldA(const _Float16* __restrict__ row, int kbA) {
    v8h lo = *reinterpret_cast<const v8h*>(row + 2 * kbA);
    v8h hi = *reinterpret_cast<const v8h*>(row + 2 * kbA + 8);
    return __builtin_shufflevector(lo, hi, 0,1,2,3,4,5,6,7,8,9,10,11,12,13,14,15);
}
// B-fragment (32x16 f16, K x N): this lane holds column (lane&15).
// colRow points at B[:,n] contiguous in K. kbB = (lane>>4)*16. One 32B load.
__device__ __forceinline__ v16h ldB(const _Float16* __restrict__ colRow, int kbB) {
    return *reinterpret_cast<const v16h*>(colRow + kbB);
}
__device__ __forceinline__ v8f wmma_f16(v16h a, v16h b, v8f c) {
    return __builtin_amdgcn_wmma_f32_16x16x32_f16(false, a, false, b,
                                                  (short)0, c, false, false);
}

// ---------------------------------------------------------------------------
__global__ void f32_to_f16(const float* __restrict__ src,
                           _Float16* __restrict__ dst, int n) {
    int i = blockIdx.x * blockDim.x + threadIdx.x;
    int stride = gridDim.x * blockDim.x;
    for (; i < n; i += stride) dst[i] = (_Float16)src[i];
}

// ---------------------------------------------------------------------------
// Fused QKV projection + RoPE.  grid = (BS/128, NH), block = 256 (8 waves).
// Each wave: 16 rows x 64 cols of q, k AND v (12 accumulators). One A-fragment
// feeds 12 WMMAs. A double-buffered across the k-loop.
//   q: RoPE, *0.125 -> [b,h,s,64]    k: RoPE -> [b,h,s,64]
//   v: transposed -> [b,h,64,s]
__global__ __launch_bounds__(256) void qkv_rope(
    const _Float16* __restrict__ xh,
    const _Float16* __restrict__ wqh, const _Float16* __restrict__ wkh,
    const _Float16* __restrict__ wvh,
    _Float16* __restrict__ Q, _Float16* __restrict__ K,
    _Float16* __restrict__ VT) {
    const int lane = threadIdx.x & 31;
    const int wave = threadIdx.x >> 5;
    const int colq = lane & 15;
    const int hi   = lane >> 4;
    const int kbA  = hi * 8, kbB = hi * 16;
    const int m0   = blockIdx.x * 128 + wave * 16;   // row in [0, BS)
    const int h    = blockIdx.y;                     // head == 64-col tile

    const _Float16* const Wm[3] = { wqh, wkh, wvh };
    v8f acc[3][4] = {};

    const _Float16* arow = xh + (size_t)(m0 + colq) * DMODEL;
    v16h a_cur = ldA(arow, kbA);
    for (int k = 0; k < DMODEL; k += 32) {
        // issue next A load first so it overlaps this iteration's WMMAs
        int kn = (k + 32 < DMODEL) ? (k + 32) : k;   // clamped, always valid
        v16h a_next = ldA(arow + kn, kbA);
#pragma unroll
        for (int z = 0; z < 3; ++z) {
            const _Float16* wz = Wm[z] + (size_t)(h * 64 + colq) * DMODEL + k;
            v16h b0 = ldB(wz,                       kbB);
            v16h b1 = ldB(wz + (size_t)16 * DMODEL, kbB);
            v16h b2 = ldB(wz + (size_t)32 * DMODEL, kbB);
            v16h b3 = ldB(wz + (size_t)48 * DMODEL, kbB);
            acc[z][0] = wmma_f16(a_cur, b0, acc[z][0]);
            acc[z][1] = wmma_f16(a_cur, b1, acc[z][1]);
            acc[z][2] = wmma_f16(a_cur, b2, acc[z][2]);
            acc[z][3] = wmma_f16(a_cur, b3, acc[z][3]);
        }
        a_cur = a_next;
    }

    // epilogue: RoPE (q,k) + layout stores
    const float LN1E4_OVER_32 = 0.28782313662425574f; // ln(10000)/32
#pragma unroll
    for (int z = 0; z < 3; ++z) {
#pragma unroll
        for (int n = 0; n < 4; ++n) {
#pragma unroll
            for (int r = 0; r < 8; ++r) {
                int gr = m0 + r + hi * 8;
                int b  = gr >> 11;            // / SEQL
                int s  = gr & (SEQL - 1);
                int hd = n * 16 + colq;
                float v = acc[z][n][r];
                if (z <= 1) {
                    float other = __shfl_xor(v, 1, 32);   // pair partner hd^1
                    int   i2    = hd >> 1;
                    float freq  = __expf(-(float)i2 * LN1E4_OVER_32);
                    float ang   = (float)s * freq;
                    float cs = __cosf(ang), sn = __sinf(ang);
                    float res = (hd & 1) ? (other * sn + v * cs)
                                         : (v * cs - other * sn);
                    if (z == 0) res *= 0.125f;            // 1/sqrt(HD)
                    _Float16* dst = (z == 0) ? Q : K;
                    dst[(((size_t)b * NH + h) * SEQL + s) * HD + hd] =
                        (_Float16)res;
                } else {
                    VT[(((size_t)b * NH + h) * HD + hd) * SEQL + s] =
                        (_Float16)v;
                }
            }
        }
    }
}

// ---------------------------------------------------------------------------
// Causal flash attention. grid = (B*NH, SEQ/128), block = 256 (8 waves).
// Each wave: 16-row q tile, online softmax, WMMA for QK^T and PV.
// K fragments double-buffered; V fragments loaded before the softmax VALU
// block so PV WMMAs never stall; prefetch two tiles ahead.
// LDS only for the P C->A relayout (per-wave region, padded stride 40).
__global__ __launch_bounds__(256) void flash_attn(
    const _Float16* __restrict__ Q, const _Float16* __restrict__ K,
    const _Float16* __restrict__ VT, _Float16* __restrict__ AO) {
    __shared__ _Float16 plds[8][16 * 40];
    const int lane = threadIdx.x & 31;
    const int wave = threadIdx.x >> 5;
    const int colq = lane & 15;
    const int hi   = lane >> 4;
    const int kbA  = hi * 8, kbB = hi * 16;
    const int b    = blockIdx.x / NH;
    const int h    = blockIdx.x % NH;
    const int q0   = blockIdx.y * 128 + wave * 16;

    const _Float16* Qb = Q  + ((size_t)b * NH + h) * SEQL * HD;
    const _Float16* Kb = K  + ((size_t)b * NH + h) * SEQL * HD;
    const _Float16* Vb = VT + ((size_t)b * NH + h) * HD * SEQL;

    // Q tile resident in registers: two A-fragments covering HD=64.
    const _Float16* qrow = Qb + (size_t)(q0 + colq) * HD;
    v16h qa0 = ldA(qrow, kbA);
    v16h qa1 = ldA(qrow + 32, kbA);

    v8f acc[4] = {};
    float m[8], l[8];
#pragma unroll
    for (int r = 0; r < 8; ++r) { m[r] = -__builtin_inff(); l[r] = 0.0f; }

    _Float16* P = plds[wave];
    const int kend = q0 + 16;                 // keys < kend (causal)

    // preload K fragments for kb = 0
    v16h kf0 = ldB(Kb + (size_t)colq * HD,        kbB);
    v16h kf1 = ldB(Kb + (size_t)colq * HD + 32,   kbB);
    v16h kf2 = ldB(Kb + (size_t)(16 + colq) * HD,      kbB);
    v16h kf3 = ldB(Kb + (size_t)(16 + colq) * HD + 32, kbB);

    for (int kb = 0; kb < kend; kb += 32) {
        // --- scores for keys kb..kb+31 ---
        v8f s0 = {}, s1 = {};
        s0 = wmma_f16(qa0, kf0, s0);
        s0 = wmma_f16(qa1, kf1, s0);
        s1 = wmma_f16(qa0, kf2, s1);
        s1 = wmma_f16(qa1, kf3, s1);

        // --- issue loads that the rest of this iteration will hide ---
        // V fragments for the current tile (used by PV after softmax)
        v16h vf0 = ldB(Vb + (size_t)(colq)      * SEQL + kb, kbB);
        v16h vf1 = ldB(Vb + (size_t)(16 + colq) * SEQL + kb, kbB);
        v16h vf2 = ldB(Vb + (size_t)(32 + colq) * SEQL + kb, kbB);
        v16h vf3 = ldB(Vb + (size_t)(48 + colq) * SEQL + kb, kbB);
        // next K fragments (clamped pointer keeps the load always valid)
        int kb2 = (kb + 32 < kend) ? (kb + 32) : kb;
        kf0 = ldB(Kb + (size_t)(kb2 + colq) * HD,           kbB);
        kf1 = ldB(Kb + (size_t)(kb2 + colq) * HD + 32,      kbB);
        kf2 = ldB(Kb + (size_t)(kb2 + 16 + colq) * HD,      kbB);
        kf3 = ldB(Kb + (size_t)(kb2 + 16 + colq) * HD + 32, kbB);
        // prefetch two tiles ahead (global_prefetch path, no LOADcnt)
        if (kb + 64 < kend) {
            __builtin_prefetch(Kb + (size_t)(kb + 64 + colq) * HD, 0, 0);
            __builtin_prefetch(Vb + (size_t)colq * SEQL + kb + 64, 0, 0);
        }

        // --- online softmax (VALU block overlaps the loads above) ---
#pragma unroll
        for (int r = 0; r < 8; ++r) {
            int row = q0 + r + hi * 8;
            float v0 = (kb + colq      <= row) ? s0[r] : -__builtin_inff();
            float v1 = (kb + 16 + colq <= row) ? s1[r] : -__builtin_inff();
            float mx = fmaxf(v0, v1);
            mx = fmaxf(mx, __shfl_xor(mx, 1, 32));
            mx = fmaxf(mx, __shfl_xor(mx, 2, 32));
            mx = fmaxf(mx, __shfl_xor(mx, 4, 32));
            mx = fmaxf(mx, __shfl_xor(mx, 8, 32));
            float mnew = fmaxf(m[r], mx);
            float p0 = __expf(v0 - mnew);            // exp(-inf)=0 masks
            float p1 = __expf(v1 - mnew);
            float rs = p0 + p1;
            rs += __shfl_xor(rs, 1, 32);
            rs += __shfl_xor(rs, 2, 32);
            rs += __shfl_xor(rs, 4, 32);
            rs += __shfl_xor(rs, 8, 32);
            float corr = __expf(m[r] - mnew);
            l[r] = l[r] * corr + rs;
            m[r] = mnew;
#pragma unroll
            for (int n = 0; n < 4; ++n) acc[n][r] *= corr;
            int prow = r + hi * 8;
            P[prow * 40 + colq]      = (_Float16)p0;
            P[prow * 40 + 16 + colq] = (_Float16)p1;
        }
        // read P back in A-fragment layout (same-wave DS ops are in-order)
        v16h pa = ldA(P + colq * 40, kbA);
        acc[0] = wmma_f16(pa, vf0, acc[0]);
        acc[1] = wmma_f16(pa, vf1, acc[1]);
        acc[2] = wmma_f16(pa, vf2, acc[2]);
        acc[3] = wmma_f16(pa, vf3, acc[3]);
    }

    // normalize and store attention output as f16 [BS, DMODEL]
#pragma unroll
    for (int n = 0; n < 4; ++n) {
#pragma unroll
        for (int r = 0; r < 8; ++r) {
            int row = q0 + r + hi * 8;
            float o = acc[n][r] / l[r];
            AO[((size_t)b * SEQL + row) * DMODEL + h * 64 + n * 16 + colq] =
                (_Float16)o;
        }
    }
}

// ---------------------------------------------------------------------------
// Output projection: out = AO @ wo^T, f32 result.
// grid = (BS/128, DMODEL/64), block = 256. A and B double-buffered.
__global__ __launch_bounds__(256) void out_proj(
    const _Float16* __restrict__ AO, const _Float16* __restrict__ woh,
    float* __restrict__ out) {
    const int lane = threadIdx.x & 31;
    const int wave = threadIdx.x >> 5;
    const int colq = lane & 15;
    const int hi   = lane >> 4;
    const int kbA  = hi * 8, kbB = hi * 16;
    const int m0   = blockIdx.x * 128 + wave * 16;
    const int n0   = blockIdx.y * 64;

    v8f acc[4] = {};
    const _Float16* arow = AO  + (size_t)(m0 + colq) * DMODEL;
    const _Float16* wrow = woh + (size_t)(n0 + colq) * DMODEL;

    v16h a_cur  = ldA(arow, kbA);
    v16h b_cur0 = ldB(wrow,                       kbB);
    v16h b_cur1 = ldB(wrow + (size_t)16 * DMODEL, kbB);
    v16h b_cur2 = ldB(wrow + (size_t)32 * DMODEL, kbB);
    v16h b_cur3 = ldB(wrow + (size_t)48 * DMODEL, kbB);

    for (int k = 0; k < DMODEL; k += 32) {
        int kn = (k + 32 < DMODEL) ? (k + 32) : k;    // clamped tail
        v16h a_next  = ldA(arow + kn, kbA);
        v16h b_next0 = ldB(wrow + kn,                       kbB);
        v16h b_next1 = ldB(wrow + kn + (size_t)16 * DMODEL, kbB);
        v16h b_next2 = ldB(wrow + kn + (size_t)32 * DMODEL, kbB);
        v16h b_next3 = ldB(wrow + kn + (size_t)48 * DMODEL, kbB);

        acc[0] = wmma_f16(a_cur, b_cur0, acc[0]);
        acc[1] = wmma_f16(a_cur, b_cur1, acc[1]);
        acc[2] = wmma_f16(a_cur, b_cur2, acc[2]);
        acc[3] = wmma_f16(a_cur, b_cur3, acc[3]);

        a_cur = a_next;
        b_cur0 = b_next0; b_cur1 = b_next1; b_cur2 = b_next2; b_cur3 = b_next3;
    }
#pragma unroll
    for (int n = 0; n < 4; ++n)
#pragma unroll
        for (int r = 0; r < 8; ++r)
            out[(size_t)(m0 + r + hi * 8) * DMODEL + n0 + n * 16 + colq] =
                acc[n][r];
}

// ---------------------------------------------------------------------------
extern "C" void kernel_launch(void* const* d_in, const int* in_sizes, int n_in,
                              void* d_out, int out_size, void* d_ws,
                              size_t ws_size, hipStream_t stream) {
    (void)in_sizes; (void)n_in; (void)out_size; (void)ws_size;
    const float* x  = (const float*)d_in[0];
    const float* wq = (const float*)d_in[1];
    const float* wk = (const float*)d_in[2];
    const float* wv = (const float*)d_in[3];
    const float* wo = (const float*)d_in[4];
    float* out = (float*)d_out;

    _Float16* ws = (_Float16*)d_ws;
    size_t o = 0;
    _Float16* xh  = ws + o; o += (size_t)BS * DMODEL;
    _Float16* wqh = ws + o; o += (size_t)DMODEL * DMODEL;
    _Float16* wkh = ws + o; o += (size_t)DMODEL * DMODEL;
    _Float16* wvh = ws + o; o += (size_t)DMODEL * DMODEL;
    _Float16* woh = ws + o; o += (size_t)DMODEL * DMODEL;
    _Float16* Qd  = ws + o; o += (size_t)BS * HD * NH;   // [b,h,s,64]
    _Float16* Kd  = ws + o; o += (size_t)BS * HD * NH;   // [b,h,s,64]
    _Float16* VTd = ws + o; o += (size_t)BS * HD * NH;   // [b,h,64,s]
    _Float16* AO  = ws + o; o += (size_t)BS * DMODEL;    // [bs, 768]

    f32_to_f16<<<2048, 256, 0, stream>>>(x,  xh,  BS * DMODEL);
    f32_to_f16<<<256,  256, 0, stream>>>(wq, wqh, DMODEL * DMODEL);
    f32_to_f16<<<256,  256, 0, stream>>>(wk, wkh, DMODEL * DMODEL);
    f32_to_f16<<<256,  256, 0, stream>>>(wv, wvh, DMODEL * DMODEL);
    f32_to_f16<<<256,  256, 0, stream>>>(wo, woh, DMODEL * DMODEL);

    dim3 g1(BS / 128, NH);
    qkv_rope<<<g1, 256, 0, stream>>>(xh, wqh, wkh, wvh, Qd, Kd, VTd);

    dim3 g2(BATCH * NH, SEQL / 128);
    flash_attn<<<g2, 256, 0, stream>>>(Qd, Kd, VTd, AO);

    dim3 g3(BS / 128, DMODEL / 64);
    out_proj<<<g3, 256, 0, stream>>>(AO, woh, out);
}
